// SEQ_VAE_45363444580939
// MI455X (gfx1250) — matrix operand order repaired and checked by reading, not compiled, not run on hardware
//
#include <hip/hip_runtime.h>
#include <hip/hip_bf16.h>

typedef __attribute__((ext_vector_type(16))) __bf16 v16bf;
typedef __attribute__((ext_vector_type(8)))  __bf16 bf16x8;
typedef __attribute__((ext_vector_type(8)))  float  v8f;

constexpr int B_  = 128;
constexpr int T_  = 256;
constexpr int DIN = 512;
constexpr int H_  = 1024;
constexpr int S_  = 128;
constexpr int DH_ = 1024;
constexpr int N_G = 4 * H_;        // 4096 gate columns
constexpr int K_G = S_ + H_;       // 1152 gate reduction dim
constexpr int K_P = DIN + H_;      // 1536 posterior reduction dim

// ---------------- WMMA helpers ----------------

__device__ inline v8f wmma_bf16(v16bf a, v16bf b, v8f c) {
  return __builtin_amdgcn_wmma_f32_16x16x32_bf16(
      /*neg_a=*/false, a, /*neg_b=*/false, b,
      /*c_mod=*/(short)0, c, /*reuse_a=*/false, /*reuse_b=*/false);
}

// A fragment (16x32 bf16, row m = lane%16):
// lane half h: VGPR0-3 hold K = h*8 + 0..7, VGPR4-7 hold K = 16 + h*8 + 0..7
__device__ inline v16bf frag_a(const __bf16* base, long ld, int lane) {
  const __bf16* p = base + (long)(lane & 15) * ld + ((lane >> 4) << 3);
  bf16x8 lo = *(const bf16x8*)(p);
  bf16x8 hi = *(const bf16x8*)(p + 16);
  v16bf r;
#pragma unroll
  for (int i = 0; i < 8; ++i) { r[i] = lo[i]; r[i + 8] = hi[i]; }
  return r;
}

// B fragment (32x16 bf16, col n = lane%16; W row-major [N][K], a B "column"
// is a W row): lane half h holds K = h*8 + 0..7, then K = 16 + h*8 + 0..7
__device__ inline v16bf frag_b(const __bf16* base, long ld, int lane) {
  const __bf16* p = base + (long)(lane & 15) * ld + ((lane >> 4) << 3);
  bf16x8 lo = *(const bf16x8*)(p);
  bf16x8 hi = *(const bf16x8*)(p + 16);
  v16bf r;
#pragma unroll
  for (int i = 0; i < 8; ++i) { r[i] = lo[i]; r[i + 8] = hi[i]; }
  return r;
}

// ---------------- prep kernels ----------------

__global__ void cvt_bf16(const float* __restrict__ src, __bf16* __restrict__ dst, int n) {
  int i = blockIdx.x * 256 + threadIdx.x;
  if (i < n) dst[i] = (__bf16)src[i];
}

// Wg[n][0:128] = W_ih[n], Wg[n][128:1152] = W_hh[n]
__global__ void pack_gate_w(const float* __restrict__ wih, const float* __restrict__ whh,
                            __bf16* __restrict__ wg) {
  long i = (long)blockIdx.x * 256 + threadIdx.x;
  if (i >= (long)N_G * K_G) return;
  int n = (int)(i / K_G), k = (int)(i % K_G);
  float v = (k < S_) ? wih[(long)n * S_ + k] : whh[(long)n * H_ + (k - S_)];
  wg[i] = (__bf16)v;
}

__global__ void init_state(float* __restrict__ c, __bf16* __restrict__ hb,
                           __bf16* __restrict__ z0) {
  int i = blockIdx.x * 256 + threadIdx.x;
  if (i < B_ * H_) { c[i] = 0.0f; hb[i] = (__bf16)0.0f; }
  if (i < B_ * S_) z0[i] = (__bf16)0.0f;
}

// ---------------- fused recurrent step: gates GEMM + LSTM cell ----------------

__device__ inline float sigf(float x) { return 1.0f / (1.0f + expf(-x)); }

// One WG owns rows m0..m0+15 (batch) x h-cols n0..n0+63, computes ALL four
// gate tiles with WMMA, exchanges f32 tiles via LDS, then applies the cell
// update in-place. grid (B/16, H/64), block 256 (8 waves):
//   wave w -> gate g = w>>1, col pair (w&1)*32 (two 16-wide tiles, shared A)
__global__ __launch_bounds__(256) void lstm_step(
    const __bf16* __restrict__ zb, const __bf16* __restrict__ hb_in,
    const __bf16* __restrict__ wg,
    const float* __restrict__ bih, const float* __restrict__ bhh,
    float* __restrict__ c, __bf16* __restrict__ hb_out) {
  __shared__ float sm[4 * 16 * 64];  // [gate][row][col] f32, 16 KB

  int lane = threadIdx.x & 31, w = threadIdx.x >> 5;
  int m0 = blockIdx.x * 16;
  int n0 = blockIdx.y * 64;
  int gate = w >> 1;
  int csub = (w & 1) * 32;

  v8f acc0 = {}, acc1 = {};
  const __bf16* w0 = wg + (long)(gate * H_ + n0 + csub) * K_G;
  const __bf16* w1 = w0 + (long)16 * K_G;

#pragma unroll
  for (int k = 0; k < S_; k += 32) {
    v16bf a  = frag_a(zb + (long)m0 * S_ + k, S_, lane);
    v16bf b0 = frag_b(w0 + k, K_G, lane);
    v16bf b1 = frag_b(w1 + k, K_G, lane);
    acc0 = wmma_bf16(a, b0, acc0);
    acc1 = wmma_bf16(a, b1, acc1);
  }
  for (int k = 0; k < H_; k += 32) {
    v16bf a  = frag_a(hb_in + (long)m0 * H_ + k, H_, lane);
    v16bf b0 = frag_b(w0 + S_ + k, K_G, lane);
    v16bf b1 = frag_b(w1 + S_ + k, K_G, lane);
    acc0 = wmma_bf16(a, b0, acc0);
    acc1 = wmma_bf16(a, b1, acc1);
  }

  // scatter f32 tiles into LDS: sm[((gate*16 + row)*64) + col]
  {
    int cl0 = csub + (lane & 15);
    int rl  = ((lane >> 4) << 3);
#pragma unroll
    for (int r = 0; r < 8; ++r) {
      sm[(gate * 16 + rl + r) * 64 + cl0]      = acc0[r];
      sm[(gate * 16 + rl + r) * 64 + cl0 + 16] = acc1[r];
    }
  }
  __syncthreads();

  // LSTM cell update: 16x64 elements, 4 per thread
#pragma unroll
  for (int e = 0; e < 4; ++e) {
    int idx = threadIdx.x + 256 * e;       // 0..1023
    int r = idx >> 6, cc = idx & 63;
    int b = m0 + r, j = n0 + cc;
    float xi = sm[(0 * 16 + r) * 64 + cc] + bih[j]          + bhh[j];
    float xf = sm[(1 * 16 + r) * 64 + cc] + bih[H_ + j]     + bhh[H_ + j];
    float xg = sm[(2 * 16 + r) * 64 + cc] + bih[2 * H_ + j] + bhh[2 * H_ + j];
    float xo = sm[(3 * 16 + r) * 64 + cc] + bih[3 * H_ + j] + bhh[3 * H_ + j];
    long ci = (long)b * H_ + j;
    float cn = sigf(xf) * c[ci] + sigf(xi) * tanhf(xg);
    c[ci] = cn;
    hb_out[ci] = (__bf16)(sigf(xo) * tanhf(cn));
  }
}

// ---------------- posterior: mean/logvar GEMM + reparam + KL ----------------

// grid 8, block 256 (8 waves): tile = blockIdx.x*8+w over 8x8 (m,n) tiles;
// each wave carries two accumulators (mean & log-var) sharing the A fragment.
__global__ __launch_bounds__(256) void posterior_step(
    const __bf16* __restrict__ xb, const __bf16* __restrict__ hb,
    const __bf16* __restrict__ wmb, const __bf16* __restrict__ wvb,
    const float* __restrict__ bm, const float* __restrict__ bv,
    const float* __restrict__ noise, __bf16* __restrict__ z_next,
    float* __restrict__ kl_part, int t) {
  int lane = threadIdx.x & 31, w = threadIdx.x >> 5;
  int tile = blockIdx.x * 8 + w;
  int m0 = (tile >> 3) * 16, n0 = (tile & 7) * 16;
  v8f cm = {}, cv = {};
  const __bf16* ax = xb + ((long)m0 * T_ + t) * DIN;   // x rows, stride T*DIN
  const __bf16* wm = wmb + (long)n0 * K_P;
  const __bf16* wv = wvb + (long)n0 * K_P;
  for (int k = 0; k < DIN; k += 32) {
    v16bf a  = frag_a(ax + k, (long)T_ * DIN, lane);
    v16bf b1 = frag_b(wm + k, K_P, lane);
    v16bf b2 = frag_b(wv + k, K_P, lane);
    cm = wmma_bf16(a, b1, cm);
    cv = wmma_bf16(a, b2, cv);
  }
  for (int k = 0; k < H_; k += 32) {
    v16bf a  = frag_a(hb + (long)m0 * H_ + k, H_, lane);
    v16bf b1 = frag_b(wm + DIN + k, K_P, lane);
    v16bf b2 = frag_b(wv + DIN + k, K_P, lane);
    cm = wmma_bf16(a, b1, cm);
    cv = wmma_bf16(a, b2, cv);
  }
  float kl = 0.0f;
  int n = n0 + (lane & 15);
  int rb = m0 + ((lane >> 4) << 3);
#pragma unroll
  for (int r = 0; r < 8; ++r) {
    int m = rb + r;
    float mean = cm[r] + bm[n];
    float lv   = cv[r] + bv[n];
    float sd   = expf(0.5f * lv);
    float zn   = noise[((long)m * T_ + t) * S_ + n] * sd + mean;
    z_next[(long)m * S_ + n] = (__bf16)zn;
    kl += sd * sd + mean * mean - 0.5f * lv - 0.5f;
  }
  __shared__ float red[256];
  red[threadIdx.x] = kl;
  __syncthreads();
#pragma unroll
  for (int s = 128; s > 0; s >>= 1) {
    if (threadIdx.x < s) red[threadIdx.x] += red[threadIdx.x + s];
    __syncthreads();
  }
  if (threadIdx.x == 0) kl_part[t * 8 + blockIdx.x] = red[0];
}

// ---------------- decoder ----------------

// hid = relu(Z @ Wd1^T + bd1). grid (BT/16, DH/512), block 256; wave -> 16x64.
__global__ __launch_bounds__(256) void dec1(
    const __bf16* __restrict__ z1,  // z_all + B*S (slots 1..T)
    const __bf16* __restrict__ wd1, const float* __restrict__ bd1,
    __bf16* __restrict__ hid) {
  int lane = threadIdx.x & 31, w = threadIdx.x >> 5;
  int m0 = blockIdx.x * 16;                 // flat row = b*T + t
  int n0 = blockIdx.y * 512 + w * 64;
  int b = m0 / T_, t0 = m0 % T_;            // 16 rows: same b, t0..t0+15
  const __bf16* arow = z1 + ((long)t0 * B_ + b) * S_;
  v8f acc[4] = {{}, {}, {}, {}};
#pragma unroll
  for (int k = 0; k < S_; k += 32) {
    v16bf a = frag_a(arow + k, (long)B_ * S_, lane);
#pragma unroll
    for (int j = 0; j < 4; ++j) {
      v16bf bf = frag_b(wd1 + (long)(n0 + 16 * j) * S_ + k, S_, lane);
      acc[j] = wmma_bf16(a, bf, acc[j]);
    }
  }
  int rb = ((lane >> 4) << 3);
#pragma unroll
  for (int j = 0; j < 4; ++j) {
    int n = n0 + 16 * j + (lane & 15);
#pragma unroll
    for (int r = 0; r < 8; ++r) {
      float v = acc[j][r] + bd1[n];
      v = v > 0.0f ? v : 0.0f;
      hid[(long)(m0 + rb + r) * DH_ + n] = (__bf16)v;
    }
  }
}

// x_hat = hid @ Wd2^T + bd2. grid BT/16, block 256; wave -> 16x64 (8*64=512=N).
__global__ __launch_bounds__(256) void dec2(
    const __bf16* __restrict__ hid, const __bf16* __restrict__ wd2,
    const float* __restrict__ bd2, float* __restrict__ out) {
  int lane = threadIdx.x & 31, w = threadIdx.x >> 5;
  int m0 = blockIdx.x * 16;
  int n0 = w * 64;
  v8f acc[4] = {{}, {}, {}, {}};
  for (int k = 0; k < DH_; k += 32) {
    // pull next K-step's weight tiles toward the WGP (global_prefetch_b8)
    if (k + 32 < DH_)
      __builtin_prefetch(wd2 + (long)n0 * DH_ + (lane & 15) * DH_ + k + 32, 0, 1);
    v16bf a = frag_a(hid + (long)m0 * DH_ + k, DH_, lane);
#pragma unroll
    for (int j = 0; j < 4; ++j) {
      v16bf bf = frag_b(wd2 + (long)(n0 + 16 * j) * DH_ + k, DH_, lane);
      acc[j] = wmma_bf16(a, bf, acc[j]);
    }
  }
  int rb = ((lane >> 4) << 3);
#pragma unroll
  for (int j = 0; j < 4; ++j) {
    int n = n0 + 16 * j + (lane & 15);
#pragma unroll
    for (int r = 0; r < 8; ++r)
      out[(long)(m0 + rb + r) * DIN + n] = acc[j][r] + bd2[n];
  }
}

__global__ void kl_reduce(const float* __restrict__ part, float* __restrict__ out) {
  __shared__ float red[256];
  float s = 0.0f;
  for (int i = threadIdx.x; i < T_ * 8; i += 256) s += part[i];
  red[threadIdx.x] = s;
  __syncthreads();
#pragma unroll
  for (int st = 128; st > 0; st >>= 1) {
    if (threadIdx.x < st) red[threadIdx.x] += red[threadIdx.x + st];
    __syncthreads();
  }
  if (threadIdx.x == 0) out[(long)B_ * T_ * DIN] = red[0];
}

// ---------------- launch ----------------

extern "C" void kernel_launch(void* const* d_in, const int* in_sizes, int n_in,
                              void* d_out, int out_size, void* d_ws, size_t ws_size,
                              hipStream_t stream) {
  (void)in_sizes; (void)n_in; (void)out_size; (void)ws_size;
  const float* x     = (const float*)d_in[0];
  const float* noise = (const float*)d_in[1];
  const float* W_ih  = (const float*)d_in[2];
  const float* W_hh  = (const float*)d_in[3];
  const float* b_ih  = (const float*)d_in[4];
  const float* b_hh  = (const float*)d_in[5];
  const float* Wm    = (const float*)d_in[6];
  const float* bm    = (const float*)d_in[7];
  const float* Wv    = (const float*)d_in[8];
  const float* bv    = (const float*)d_in[9];
  const float* Wd1   = (const float*)d_in[10];
  const float* bd1   = (const float*)d_in[11];
  const float* Wd2   = (const float*)d_in[12];
  const float* bd2   = (const float*)d_in[13];
  float* out = (float*)d_out;

  char* p = (char*)d_ws;
  auto alloc = [&](size_t bytes) -> char* {
    char* r = p;
    p += (bytes + 255) & ~(size_t)255;
    return r;
  };
  __bf16* xbf   = (__bf16*)alloc((size_t)B_ * T_ * DIN * 2);
  __bf16* wg    = (__bf16*)alloc((size_t)N_G * K_G * 2);
  __bf16* wmb   = (__bf16*)alloc((size_t)S_ * K_P * 2);
  __bf16* wvb   = (__bf16*)alloc((size_t)S_ * K_P * 2);
  __bf16* wd1b  = (__bf16*)alloc((size_t)DH_ * S_ * 2);
  __bf16* wd2b  = (__bf16*)alloc((size_t)DIN * DH_ * 2);
  __bf16* hb    = (__bf16*)alloc((size_t)B_ * H_ * 2);
  float*  cst   = (float*)alloc((size_t)B_ * H_ * 4);
  __bf16* z_all = (__bf16*)alloc((size_t)(T_ + 1) * B_ * S_ * 2);
  __bf16* hid   = (__bf16*)alloc((size_t)B_ * T_ * DH_ * 2);
  float*  klp   = (float*)alloc((size_t)T_ * 8 * 4);

  int n;
  n = B_ * T_ * DIN; cvt_bf16<<<(n + 255) / 256, 256, 0, stream>>>(x,   xbf,  n);
  n = S_ * K_P;      cvt_bf16<<<(n + 255) / 256, 256, 0, stream>>>(Wm,  wmb,  n);
  n = S_ * K_P;      cvt_bf16<<<(n + 255) / 256, 256, 0, stream>>>(Wv,  wvb,  n);
  n = DH_ * S_;      cvt_bf16<<<(n + 255) / 256, 256, 0, stream>>>(Wd1, wd1b, n);
  n = DIN * DH_;     cvt_bf16<<<(n + 255) / 256, 256, 0, stream>>>(Wd2, wd2b, n);
  {
    long tot = (long)N_G * K_G;
    pack_gate_w<<<(unsigned)((tot + 255) / 256), 256, 0, stream>>>(W_ih, W_hh, wg);
  }
  init_state<<<(B_ * H_ + 255) / 256, 256, 0, stream>>>(cst, hb, z_all);

  for (int t = 0; t < T_; ++t) {
    lstm_step<<<dim3(B_ / 16, H_ / 64), 256, 0, stream>>>(
        z_all + (size_t)t * B_ * S_, hb, wg, b_ih, b_hh, cst, hb);
    posterior_step<<<8, 256, 0, stream>>>(
        xbf, hb, wmb, wvb, bm, bv, noise,
        z_all + (size_t)(t + 1) * B_ * S_, klp, t);
  }

  dec1<<<dim3(B_ * T_ / 16, DH_ / 512), 256, 0, stream>>>(
      z_all + (size_t)B_ * S_, wd1b, bd1, hid);
  dec2<<<dim3(B_ * T_ / 16), 256, 0, stream>>>(hid, wd2b, bd2, out);
  kl_reduce<<<1, 256, 0, stream>>>(klp, out);
}
